// CubeNetOLD_44564580663882
// MI455X (gfx1250) — matrix operand order repaired and checked by reading, not compiled, not run on hardware
//
#include <hip/hip_runtime.h>
#include <hip/hip_fp16.h>

#define B_      16
#define Hn      500
#define HP      512
#define T_      2048
#define NWG     32
#define TPB     128
#define CDIM    256
#define MGC     80
#define NFR     32

typedef __attribute__((ext_vector_type(16))) _Float16 v16h;
typedef __attribute__((ext_vector_type(8)))  float    v8f;

// workspace layout (bytes)
#define WS_SYNC    0          // int counter
#define WS_H       1024       // 2 * [16][512] f16  = 32768 B (double buffered h)
#define WS_PART    34816      // 2 * [2][32][16] f32 = 8192 B (double buffered partials)
#define WS_ZERO    65536      // memset range covering the above
#define WS_COND    65536      // [512][256] f32 = 524288 B
#define WS_CG      1048576    // [32][16][2048] f32 = 4 MB

// ---- precompute 1: cond[b*32+f][c] = mgc[b,f,:] @ W_up + b_up --------------
__global__ void cond_proj_kernel(const float* __restrict__ mgc,
                                 const float* __restrict__ W_up,
                                 const float* __restrict__ b_up,
                                 float* __restrict__ cond) {
  const int bf = blockIdx.x;        // 0..511  (b*32 + f)
  const int c  = threadIdx.x;       // 0..255
  const float* m = mgc + bf * MGC;
  float acc = b_up[c];
#pragma unroll 8
  for (int k = 0; k < MGC; ++k) acc += m[k] * W_up[k * CDIM + c];
  cond[bf * CDIM + c] = acc;
}

// ---- precompute 2: cg[f][b][jpad] = cond @ W_ih[:, :256]^T + b_ih + b_hh ----
// padded gate space: 4 chunks (i,f,g,o) of 512; entries with jj>=500 are 0.
__global__ void cond_gates_kernel(const float* __restrict__ cond,
                                  const float* __restrict__ W_ih,
                                  const float* __restrict__ b_ih,
                                  const float* __restrict__ b_hh,
                                  float* __restrict__ cg) {
  const int gid = blockIdx.x * blockDim.x + threadIdx.x;  // 0 .. 32*16*2048-1
  const int j  = gid & 2047;
  const int fb = gid >> 11;       // f*16 + b
  const int f  = fb >> 4;
  const int b  = fb & 15;
  const int w  = j >> 9;          // gate chunk 0..3
  const int jj = j & 511;
  float acc = 0.f;
  if (jj < Hn) {
    const int jr = w * Hn + jj;                 // real gate row 0..1999
    acc = b_ih[jr] + b_hh[jr];
    const float* cv = cond + (b * NFR + f) * CDIM;
    const float* wr = W_ih + jr * 257;
#pragma unroll 8
    for (int c2 = 0; c2 < CDIM; ++c2) acc += cv[c2] * wr[c2];
  }
  cg[gid] = acc;
}

// ---- persistent autoregressive LSTM kernel ---------------------------------
__global__ __launch_bounds__(TPB, 1)
void wavernn_persist(const float* __restrict__ eps,
                     const float* __restrict__ W_ih,
                     const float* __restrict__ W_hh,
                     const float* __restrict__ W_out,
                     const float* __restrict__ b_out,
                     const float* __restrict__ cg,
                     _Float16* __restrict__ hbuf,   // 2 x [16][512]
                     float* __restrict__ part,      // 2 x [2][32][16]
                     int* __restrict__ sync_cnt,
                     float* __restrict__ out) {
  const int tid = threadIdx.x;
  const int wv  = tid >> 5;      // wave = gate chunk (i,f,g,o)
  const int ln  = tid & 31;
  const int g   = blockIdx.x;    // WG owns h rows [16g, 16g+16)

  __shared__ float gate_lds[4][8][32];
  __shared__ float h_lds[16][16];    // [k][b]
  __shared__ float zz_lds[16];
  __shared__ float wo_lds[2][16];

  if (tid < 16) {
    const int kr = g * 16 + tid;
    wo_lds[0][tid] = (kr < Hn) ? W_out[kr] : 0.f;
    wo_lds[1][tid] = (kr < Hn) ? W_out[Hn + kr] : 0.f;
  }

  // --- stationary W_hh slab as f16 WMMA B-fragments (16 k-blocks of 32) ---
  const int  nloc = g * 16 + (ln & 15);       // local gate row within chunk
  const bool nval = (nloc < Hn);
  const int  nrow = wv * Hn + nloc;           // row of W_hh (2000 x 500)
  const int  koff = (ln >= 16) ? 16 : 0;      // B layout: lanes 0-15 K=0..7,16..23; 16-31 K=8..15,24..31
  v16h Bf[16];
#pragma unroll
  for (int kb = 0; kb < 16; ++kb) {
#pragma unroll
    for (int v = 0; v < 8; ++v) {
      const int k0 = kb * 32 + hiB + 2 * v + ((v >= 4) ? 8 : 0);
      const float a0 = (nval && k0     < Hn) ? W_hh[nrow * Hn + k0    ] : 0.f;
      const float a1 = (nval && k0 + 1 < Hn) ? W_hh[nrow * Hn + k0 + 1] : 0.f;
      Bf[kb][2 * v    ] = (_Float16)a0;
      Bf[kb][2 * v + 1] = (_Float16)a1;
    }
  }

  const float wx = nval ? W_ih[nrow * 257 + 256] : 0.f;  // x_prev weight

  // A-fragment addressing: lane = batch, two 16B runs per 32-k block
  const int bA   = ln & 15;
  const int selA = (ln >= 16) ? 8 : 0;
  const int hiB  = (ln >= 16) ? 8 : 0;
  const int jg   = wv * 512 + g * 16 + (ln & 15);  // padded gate index for D col

  // elementwise assignment: 2 (k,b) cells per thread
  const int k0e = tid >> 4, k1e = 8 + (tid >> 4), be = tid & 15;
  float c0 = 0.f, c1 = 0.f;
  float cnd[8];

  for (int t = 0; t < T_; ++t) {
    if ((t & 63) == 0) {                      // new conditioning frame
      const int f = t >> 6;
#pragma unroll
      for (int r = 0; r < 8; ++r)
        cnd[r] = cg[(f * 16 + (r + hiB)) * 2048 + jg];
    }

    // reconstruct zz_{t-1} from step t-1 partials (every WG, redundantly)
    if (tid < 16) {
      float zzv = 0.f;
      if (t > 0) {
        const float* p = part + ((t - 1) & 1) * 1024;
        float s0 = 0.f, s1 = 0.f;
#pragma unroll 8
        for (int gg = 0; gg < NWG; ++gg) {
          s0 += p[gg * 16 + tid];
          s1 += p[512 + gg * 16 + tid];
        }
        const float mean = tanhf(s0 + b_out[0]);
        const float lv   = fmaxf(s1 + b_out[1], -7.0f);
        zzv = mean + eps[tid * T_ + (t - 1)] * __expf(lv);
        if (g == 0) {
          out[             tid * T_ + (t - 1)] = mean;
          out[    B_ * T_ + tid * T_ + (t - 1)] = lv;
          out[2 * B_ * T_ + tid * T_ + (t - 1)] = zzv;
        }
      }
      zz_lds[tid] = zzv;
    }
    __syncthreads();

    // gates tile = h_{t-1} @ W_hh^T  (16 WMMAs, K=512)
    const _Float16* hsrc = hbuf + ((t + 1) & 1) * (B_ * HP);
    v8f acc = {0.f, 0.f, 0.f, 0.f, 0.f, 0.f, 0.f, 0.f};
#pragma unroll
    for (int kb = 0; kb < 16; ++kb) {
      union { v16h h; float4 q[2]; } A;
      const _Float16* ap = hsrc + bA * HP + kb * 32 + selA;
      A.q[0] = *(const float4*)(ap);
      A.q[1] = *(const float4*)(ap + 16);
      acc = __builtin_amdgcn_wmma_f32_16x16x32_f16(false, A.h, false, Bf[kb],
                                                   (short)0, acc, false, false);
    }
#pragma unroll
    for (int r = 0; r < 8; ++r)
      acc[r] += cnd[r] + zz_lds[r + hiB] * wx;
#pragma unroll
    for (int r = 0; r < 8; ++r) gate_lds[wv][r][ln] = acc[r];
    __syncthreads();

    // LSTM elementwise for this WG's h slab; c lives in registers
    _Float16* hdst = hbuf + (t & 1) * (B_ * HP);
    {
      const int li = k0e + ((be >> 3) << 4), rr = be & 7;
      float iv = gate_lds[0][rr][li], fv = gate_lds[1][rr][li];
      float gv = gate_lds[2][rr][li], ov = gate_lds[3][rr][li];
      iv = 1.f / (1.f + __expf(-iv));
      fv = 1.f / (1.f + __expf(-fv));
      gv = tanhf(gv);
      ov = 1.f / (1.f + __expf(-ov));
      c0 = fv * c0 + iv * gv;
      const float hv = ov * tanhf(c0);
      h_lds[k0e][be] = hv;
      hdst[be * HP + g * 16 + k0e] = (_Float16)hv;
    }
    {
      const int li = k1e + ((be >> 3) << 4), rr = be & 7;
      float iv = gate_lds[0][rr][li], fv = gate_lds[1][rr][li];
      float gv = gate_lds[2][rr][li], ov = gate_lds[3][rr][li];
      iv = 1.f / (1.f + __expf(-iv));
      fv = 1.f / (1.f + __expf(-fv));
      gv = tanhf(gv);
      ov = 1.f / (1.f + __expf(-ov));
      c1 = fv * c1 + iv * gv;
      const float hv = ov * tanhf(c1);
      h_lds[k1e][be] = hv;
      hdst[be * HP + g * 16 + k1e] = (_Float16)hv;
    }
    __syncthreads();

    // partial output projection for this slab
    if (tid < 16) {
      float s0 = 0.f, s1 = 0.f;
#pragma unroll
      for (int k = 0; k < 16; ++k) {
        const float hv = h_lds[k][tid];
        s0 += hv * wo_lds[0][k];
        s1 += hv * wo_lds[1][k];
      }
      float* p = part + (t & 1) * 1024;
      p[g * 16 + tid]       = s0;
      p[512 + g * 16 + tid] = s1;
    }

    // grid barrier (release -> arrive -> acquire)
    __threadfence();
    __syncthreads();
    if (tid == 0) {
      __hip_atomic_fetch_add(sync_cnt, 1, __ATOMIC_ACQ_REL, __HIP_MEMORY_SCOPE_AGENT);
      const int target = NWG * (t + 1);
      while (__hip_atomic_load(sync_cnt, __ATOMIC_ACQUIRE, __HIP_MEMORY_SCOPE_AGENT) < target)
        __builtin_amdgcn_s_sleep(1);
    }
    __syncthreads();
    __threadfence();
  }

  // final column t = T-1
  if (g == 0 && tid < 16) {
    const float* p = part + ((T_ - 1) & 1) * 1024;
    float s0 = 0.f, s1 = 0.f;
#pragma unroll 8
    for (int gg = 0; gg < NWG; ++gg) {
      s0 += p[gg * 16 + tid];
      s1 += p[512 + gg * 16 + tid];
    }
    const float mean = tanhf(s0 + b_out[0]);
    const float lv   = fmaxf(s1 + b_out[1], -7.0f);
    const float zzv  = mean + eps[tid * T_ + (T_ - 1)] * __expf(lv);
    out[             tid * T_ + (T_ - 1)] = mean;
    out[    B_ * T_ + tid * T_ + (T_ - 1)] = lv;
    out[2 * B_ * T_ + tid * T_ + (T_ - 1)] = zzv;
  }
}

extern "C" void kernel_launch(void* const* d_in, const int* in_sizes, int n_in,
                              void* d_out, int out_size, void* d_ws, size_t ws_size,
                              hipStream_t stream) {
  (void)in_sizes; (void)n_in; (void)out_size; (void)ws_size;
  const float* mgc   = (const float*)d_in[0];
  const float* eps   = (const float*)d_in[1];
  const float* W_up  = (const float*)d_in[2];
  const float* b_up  = (const float*)d_in[3];
  const float* W_ih  = (const float*)d_in[4];
  const float* W_hh  = (const float*)d_in[5];
  const float* b_ih  = (const float*)d_in[6];
  const float* b_hh  = (const float*)d_in[7];
  const float* W_out = (const float*)d_in[8];
  const float* b_out = (const float*)d_in[9];
  float* out = (float*)d_out;
  char*  ws  = (char*)d_ws;

  // zero sync counter, h double-buffer, partial double-buffer
  hipMemsetAsync(ws, 0, WS_ZERO, stream);

  float*    cond = (float*)(ws + WS_COND);
  float*    cg   = (float*)(ws + WS_CG);
  _Float16* hbuf = (_Float16*)(ws + WS_H);
  float*    part = (float*)(ws + WS_PART);
  int*      cnt  = (int*)(ws + WS_SYNC);

  cond_proj_kernel <<<B_ * NFR, CDIM, 0, stream>>>(mgc, W_up, b_up, cond);
  cond_gates_kernel<<<(NFR * B_ * 2048) / 256, 256, 0, stream>>>(cond, W_ih, b_ih, b_hh, cg);
  wavernn_persist  <<<NWG, TPB, 0, stream>>>(eps, W_ih, W_hh, W_out, b_out,
                                             cg, hbuf, part, cnt, out);
}
